// SwinTransformerBlock_58737972740781
// MI455X (gfx1250) — matrix operand order repaired and checked by reading, not compiled, not run on hardware
//
#include <hip/hip_runtime.h>

// ---------------- problem constants ----------------
constexpr int kB      = 8;
constexpr int kC      = 192;
constexpr int kH      = 128;
constexpr int kW      = 128;
constexpr int kHW     = kH * kW;          // 16384
constexpr int kHeads  = 6;
constexpr int kWS     = 8;
constexpr int kShift  = 4;
constexpr int kN      = 64;               // tokens per window
constexpr int kHidden = 768;
constexpr int kNwx    = 16;               // windows per side
constexpr int kNwin   = kB * kNwx * kNwx; // 2048

typedef __attribute__((ext_vector_type(4)))  float  v4f;
typedef __attribute__((ext_vector_type(8)))  float  v8f;
typedef __attribute__((ext_vector_type(16))) __bf16 v16bf;

// ---------------- helpers ----------------
__device__ __forceinline__ __bf16 f2bf(float f) { return (__bf16)f; }

// pack two bf16 into one 32-bit LDS/global store (p must be 4B aligned)
__device__ __forceinline__ void store_bf2(__bf16* p, float a, float b) {
  union { __bf16 h[2]; unsigned u; } r;
  r.h[0] = (__bf16)a; r.h[1] = (__bf16)b;
  *(unsigned*)p = r.u;
}

__device__ __forceinline__ v8f vzero8() {
  v8f c;
#pragma unroll
  for (int i = 0; i < 8; ++i) c[i] = 0.0f;
  return c;
}

// A fragment: 16x32 bf16, row-major source with leading dim ld (elements).
// ISA layout: lane<16 -> K 0..7 / 16..23 ; lane>=16 -> K 8..15 / 24..31 ; M = lane&15.
__device__ __forceinline__ v16bf load_frag_a(const __bf16* p, int ld) {
  const int lane = threadIdx.x & 31;
  const int m  = lane & 15;
  const int kh = (lane & 16) ? 8 : 0;
  union { v16bf v; unsigned u[8]; } r;
  const __bf16* row = p + m * ld + kh;
#pragma unroll
  for (int j = 0; j < 8; ++j) {
    const int kb = ((j & 4) << 2) + ((j & 3) << 1); // j<4: 2j ; j>=4: 16+2(j-4)
    r.u[j] = *(const unsigned*)(row + kb);
  }
  return r.v;
}

// B fragment: 32x16 bf16 (KxN) with memory storing B^T ("(out,in)" weights):
// element(k,n) = p[n*ld + k].  N = lane&15 ; lanes<16: K 0..7 / 16..23, lanes>=16: K 8..15 / 24..31.
__device__ __forceinline__ v16bf load_frag_b(const __bf16* p, int ld) {
  const int lane = threadIdx.x & 31;
  const int n  = lane & 15;
  const int kh = (lane & 16) ? 8 : 0;
  union { v16bf v; unsigned u[8]; } r;
  const __bf16* col = p + n * ld + kh;
#pragma unroll
  for (int j = 0; j < 8; ++j) r.u[j] = *(const unsigned*)(col + ((j & 4) << 2) + ((j & 3) << 1));
  return r.v;
}

__device__ __forceinline__ v8f wmma_bf16(v16bf a, v16bf b, v8f c) {
  return __builtin_amdgcn_wmma_f32_16x16x32_bf16(false, a, false, b,
                                                 (short)0, c, false, false);
}

// shifted-window region label for token t of window (wy,wx)
__device__ __forceinline__ int win_label(int wy, int wx, int t) {
  int hs = wy * kWS + (t >> 3);
  int ws = wx * kWS + (t & 7);
  int lh = (hs < kH - kWS) ? 0 : ((hs < kH - kShift) ? 1 : 2);
  int lw = (ws < kW - kWS) ? 0 : ((ws < kW - kShift) ? 1 : 2);
  return lh * 3 + lw;
}

// ---------------- kernel: weight fp32 -> bf16 ----------------
__global__ void prep_weights(const float* qkv_w, const float* proj_w,
                             const float* fc1_w, const float* fc2_w,
                             __bf16* qkvw, __bf16* projw,
                             __bf16* fc1w, __bf16* fc2w) {
  const int i = blockIdx.x * blockDim.x + threadIdx.x;
  const int s = gridDim.x * blockDim.x;
  for (int t = i; t < 576 * 192; t += s) qkvw[t] = f2bf(qkv_w[t]);
  for (int t = i; t < 192 * 192; t += s) projw[t] = f2bf(proj_w[t]);
  for (int t = i; t < 768 * 192; t += s) fc1w[t]  = f2bf(fc1_w[t]);
  for (int t = i; t < 192 * 768; t += s) fc2w[t]  = f2bf(fc2_w[t]);
}

// ---------------- kernel: CPB MLP -> 16*sigmoid bias table (6x64x64) --------
__global__ void cpb_bias_kernel(const float* w1, const float* b1,
                                const float* w2, float* bias16) {
  __shared__ float table[225 * 6];
  const int tid = threadIdx.x;
  if (tid < 225) {
    const int a = tid / 15, b = tid % 15;
    float c0 = (float)(a - 7) / 7.0f * 8.0f;
    float c1 = (float)(b - 7) / 7.0f * 8.0f;
    const float il8 = 1.0f / 3.0f; // 1/log2(8)
    float r0 = (c0 > 0.f ? 1.f : (c0 < 0.f ? -1.f : 0.f)) * log2f(fabsf(c0) + 1.f) * il8;
    float r1 = (c1 > 0.f ? 1.f : (c1 < 0.f ? -1.f : 0.f)) * log2f(fabsf(c1) + 1.f) * il8;
    float acc[6] = {0.f, 0.f, 0.f, 0.f, 0.f, 0.f};
    for (int u = 0; u < 512; ++u) {
      float hv = fmaxf(r0 * w1[2 * u] + r1 * w1[2 * u + 1] + b1[u], 0.f);
#pragma unroll
      for (int h = 0; h < 6; ++h) acc[h] += hv * w2[h * 512 + u];
    }
#pragma unroll
    for (int h = 0; h < 6; ++h) table[tid * 6 + h] = acc[h];
  }
  __syncthreads();
  for (int idx = tid; idx < kHeads * kN * kN; idx += blockDim.x) {
    int h = idx >> 12;
    int i = (idx >> 6) & 63, j = idx & 63;
    int ridx = ((i >> 3) - (j >> 3) + 7) * 15 + ((i & 7) - (j & 7) + 7);
    float v = table[ridx * 6 + h];
    bias16[idx] = 16.0f / (1.0f + expf(-v));
  }
}

// ---------------- kernel: fused LN1 + QKV + window attention + proj ---------
// one workgroup (384 threads = 12 waves) per window
__global__ __launch_bounds__(384)
void attn_kernel(const float* __restrict__ x,
                 const float* __restrict__ n1w, const float* __restrict__ n1b,
                 const __bf16* __restrict__ qkvw,
                 const float* __restrict__ q_bias, const float* __restrict__ v_bias,
                 const float* __restrict__ logit_scale,
                 const float* __restrict__ bias16,
                 const __bf16* __restrict__ projw, const float* __restrict__ proj_b,
                 float* __restrict__ x1) {
  extern __shared__ char smem[];
  __bf16* xn     = (__bf16*)(smem + 0);      // 64x192 bf16        [0,24576)
  float*  xstage = (float*)(smem + 24576);   // 192x64 f32 stage (in qkvf region)
  float*  qkvf   = (float*)(smem + 24576);   // 64x576 f32         [24576,172032)
  __bf16* qn     = (__bf16*)(smem + 0);      // 64x192 bf16 (reuse xn)
  __bf16* kn     = (__bf16*)(smem + 172032); // 64x192 bf16
  __bf16* vt     = (__bf16*)(smem + 196608); // 192x64 bf16 (transposed v)
  float*  sS     = (float*)(smem + 24576);   // 6x64x64 f32 (reuse qkvf)
  __bf16* sP     = (__bf16*)(smem + 122880); // 6x64x64 bf16
  float*  outf   = (float*)(smem + 24576);   // 64x192 f32 (reuse sS after P)
  __bf16* outb   = (__bf16*)(smem + 73728);  // 64x192 bf16
  float*  red    = (float*)(smem + 221184);  // 2x64x6 f32 reduction scratch

  const int tid = threadIdx.x;
  const int wv  = tid >> 5;                  // wave 0..11
  const int w   = blockIdx.x;
  const int b   = w >> 8;
  const int wy  = (w >> 4) & 15;
  const int wx  = w & 15;
  const int t0  = tid / 6;                   // token 0..63
  const int g0  = tid % 6;                   // channel group 0..5

  // ---- phase 0a: coalesced stage of rolled window tile (float4 / b128) -----
  // quad layout: idx -> channel c, grp = (iy,half); 4 consecutive W pixels are
  // contiguous (shift=4 keeps quads 4-aligned mod 128, so no wrap split).
#pragma unroll
  for (int it = 0; it < 8; ++it) {
    const int idx  = it * 384 + tid;         // 0..3071
    const int c    = idx >> 4;
    const int grp  = idx & 15;
    const int iy   = grp >> 1;
    const int half = grp & 1;
    const int h0 = (wy * kWS + iy + kShift) & (kH - 1);
    const int w0 = (wx * kWS + half * 4 + kShift) & (kW - 1);
    v4f g = *(const v4f*)(x + ((size_t)(b * kC + c) * kH + h0) * kW + w0);
    *(v4f*)(xstage + c * 64 + grp * 4) = g;
  }
  __syncthreads();

  // ---- phase 0b: LN1 from LDS stage -> xn (bf16) ---------------------------
  {
    float vals[32];
    float s = 0.f, s2 = 0.f;
#pragma unroll
    for (int j = 0; j < 32; ++j) {
      float v = xstage[(g0 * 32 + j) * 64 + t0];
      vals[j] = v; s += v; s2 += v * v;
    }
    red[t0 * 6 + g0]       = s;
    red[384 + t0 * 6 + g0] = s2;
    __syncthreads();
    float ts = 0.f, ts2 = 0.f;
#pragma unroll
    for (int g = 0; g < 6; ++g) { ts += red[t0 * 6 + g]; ts2 += red[384 + t0 * 6 + g]; }
    float mu   = ts / (float)kC;
    float var  = ts2 / (float)kC - mu * mu;
    float rstd = rsqrtf(var + 1e-5f);
#pragma unroll
    for (int j = 0; j < 32; j += 2) {
      int c = g0 * 32 + j;
      store_bf2(xn + t0 * kC + c,
                (vals[j]     - mu) * rstd * n1w[c]     + n1b[c],
                (vals[j + 1] - mu) * rstd * n1w[c + 1] + n1b[c + 1]);
    }
  }
  __syncthreads();

  // ---- phase 1: QKV GEMM 64x192x576 (bf16 WMMA), + q/v bias -> qkvf --------
  {
    v8f acc[4][3];
#pragma unroll
    for (int mt = 0; mt < 4; ++mt)
#pragma unroll
      for (int nt = 0; nt < 3; ++nt) acc[mt][nt] = vzero8();
    for (int kk = 0; kk < 6; ++kk) {
      v16bf a[4];
#pragma unroll
      for (int mt = 0; mt < 4; ++mt) a[mt] = load_frag_a(xn + mt * 16 * kC + kk * 32, kC);
#pragma unroll
      for (int nt = 0; nt < 3; ++nt) {
        const int ncol = (wv * 3 + nt) * 16;
        v16bf bf = load_frag_b(qkvw + (size_t)ncol * kC + kk * 32, kC);
#pragma unroll
        for (int mt = 0; mt < 4; ++mt) acc[mt][nt] = wmma_bf16(a[mt], bf, acc[mt][nt]);
      }
    }
    const int lane = tid & 31;
    const int n    = lane & 15;
    const int mo   = (lane & 16) ? 8 : 0;
#pragma unroll
    for (int nt = 0; nt < 3; ++nt) {
      const int col = (wv * 3 + nt) * 16 + n;
      const float bias = (col < 192) ? q_bias[col] : (col >= 384 ? v_bias[col - 384] : 0.f);
#pragma unroll
      for (int mt = 0; mt < 4; ++mt)
#pragma unroll
        for (int r = 0; r < 8; ++r)
          qkvf[(mt * 16 + mo + r) * 576 + col] = acc[mt][nt][r] + bias;
    }
  }
  __syncthreads();

  // ---- phase 2: L2-normalize q,k (per token,head); transpose v -> bf16 -----
  {
    float tv[32];
    float s = 0.f;
#pragma unroll
    for (int d = 0; d < 32; ++d) { float v = qkvf[t0 * 576 + g0 * 32 + d]; tv[d] = v; s += v * v; }
    float rn = 1.0f / fmaxf(sqrtf(s), 1e-12f);
#pragma unroll
    for (int d = 0; d < 32; d += 2)
      store_bf2(qn + t0 * kC + g0 * 32 + d, tv[d] * rn, tv[d + 1] * rn);
    s = 0.f;
#pragma unroll
    for (int d = 0; d < 32; ++d) { float v = qkvf[t0 * 576 + 192 + g0 * 32 + d]; tv[d] = v; s += v * v; }
    rn = 1.0f / fmaxf(sqrtf(s), 1e-12f);
#pragma unroll
    for (int d = 0; d < 32; d += 2)
      store_bf2(kn + t0 * kC + g0 * 32 + d, tv[d] * rn, tv[d + 1] * rn);
#pragma unroll
    for (int d = 0; d < 32; ++d)
      vt[(g0 * 32 + d) * kN + t0] = f2bf(qkvf[t0 * 576 + 384 + g0 * 32 + d]);
  }
  __syncthreads();

  // ---- phase 3: S = qn*kn^T (WMMA) * exp(clamped ls) + cpb + mask ----------
  {
    const int h    = wv >> 1;
    const int half = wv & 1;
    v8f acc[2][4];
#pragma unroll
    for (int mt = 0; mt < 2; ++mt)
#pragma unroll
      for (int nt = 0; nt < 4; ++nt) acc[mt][nt] = vzero8();
    v16bf a[2];
#pragma unroll
    for (int mt = 0; mt < 2; ++mt)
      a[mt] = load_frag_a(qn + (half * 2 + mt) * 16 * kC + h * 32, kC);
#pragma unroll
    for (int nt = 0; nt < 4; ++nt) {
      v16bf bf = load_frag_b(kn + nt * 16 * kC + h * 32, kC);
#pragma unroll
      for (int mt = 0; mt < 2; ++mt) acc[mt][nt] = wmma_bf16(a[mt], bf, acc[mt][nt]);
    }
    const float ls = expf(fminf(logit_scale[h], 4.60517019f)); // log(100)
    const int lane = tid & 31;
    const int n    = lane & 15;
    const int mo   = (lane & 16) ? 8 : 0;
    const float* bh = bias16 + h * 4096;
#pragma unroll
    for (int nt = 0; nt < 4; ++nt) {
      const int col = nt * 16 + n;
      const int lc  = win_label(wy, wx, col);
#pragma unroll
      for (int mt = 0; mt < 2; ++mt)
#pragma unroll
        for (int r = 0; r < 8; ++r) {
          const int row  = (half * 2 + mt) * 16 + mo + r;
          const float mk = (win_label(wy, wx, row) != lc) ? -100.f : 0.f;
          sS[h * 4096 + row * 64 + col] = acc[mt][nt][r] * ls + bh[row * 64 + col] + mk;
        }
    }
  }
  __syncthreads();

  // ---- softmax per (head,row) -> P bf16 (exp cached in registers) ----------
  {
    const int h   = tid >> 6;   // 0..5
    const int row = tid & 63;
    const float* sr = sS + h * 4096 + row * 64;
    float e[64];
    float mx = -1e30f;
#pragma unroll
    for (int j = 0; j < 64; ++j) { e[j] = sr[j]; mx = fmaxf(mx, e[j]); }
    float sum = 0.f;
#pragma unroll
    for (int j = 0; j < 64; ++j) { e[j] = expf(e[j] - mx); sum += e[j]; }
    const float inv = 1.0f / sum;
    __bf16* pr = sP + h * 4096 + row * 64;
#pragma unroll
    for (int j = 0; j < 64; j += 2) store_bf2(pr + j, e[j] * inv, e[j + 1] * inv);
  }
  __syncthreads();

  // ---- phase 4: out = P @ v (WMMA) -> outf (64x192 f32) --------------------
  {
    const int h    = wv >> 1;
    const int half = wv & 1;
    v8f acc[2][2];
#pragma unroll
    for (int mt = 0; mt < 2; ++mt)
#pragma unroll
      for (int nt = 0; nt < 2; ++nt) acc[mt][nt] = vzero8();
    for (int kk = 0; kk < 2; ++kk) {
      v16bf a[2];
#pragma unroll
      for (int mt = 0; mt < 2; ++mt)
        a[mt] = load_frag_a(sP + h * 4096 + (half * 2 + mt) * 16 * 64 + kk * 32, 64);
#pragma unroll
      for (int nt = 0; nt < 2; ++nt) {
        v16bf bf = load_frag_b(vt + (size_t)(h * 32 + nt * 16) * kN + kk * 32, kN);
#pragma unroll
        for (int mt = 0; mt < 2; ++mt) acc[mt][nt] = wmma_bf16(a[mt], bf, acc[mt][nt]);
      }
    }
    const int lane = tid & 31;
    const int n    = lane & 15;
    const int mo   = (lane & 16) ? 8 : 0;
#pragma unroll
    for (int nt = 0; nt < 2; ++nt)
#pragma unroll
      for (int mt = 0; mt < 2; ++mt)
#pragma unroll
        for (int r = 0; r < 8; ++r) {
          const int row = (half * 2 + mt) * 16 + mo + r;
          outf[row * kC + h * 32 + nt * 16 + n] = acc[mt][nt][r];
        }
  }
  __syncthreads();

  // ---- phase 5: outf f32 -> outb bf16 (packed pair stores) -----------------
#pragma unroll
  for (int j = 0; j < 32; j += 2) {
    const int i = tid * 32 + j;
    store_bf2(outb + i, outf[i], outf[i + 1]);
  }
  __syncthreads();

  // ---- phase 6: proj GEMM + residual, reverse-roll scatter (float4) --------
  {
    v8f acc[4];
#pragma unroll
    for (int mt = 0; mt < 4; ++mt) acc[mt] = vzero8();
    for (int kk = 0; kk < 6; ++kk) {
      v16bf bf = load_frag_b(projw + (size_t)(wv * 16) * kC + kk * 32, kC);
#pragma unroll
      for (int mt = 0; mt < 4; ++mt) {
        v16bf a = load_frag_a(outb + mt * 16 * kC + kk * 32, kC);
        acc[mt] = wmma_bf16(a, bf, acc[mt]);
      }
    }
    const int lane = tid & 31;
    const int n    = lane & 15;
    const int mo   = (lane & 16) ? 8 : 0;
    const int col  = wv * 16 + n;
    const float pb = proj_b[col];
    const float* xc = x  + ((size_t)b * kC + col) * kHW;
    float*      x1c = x1 + ((size_t)b * kC + col) * kHW;
#pragma unroll
    for (int mt = 0; mt < 4; ++mt) {
      const int tb = mt * 16 + mo;                 // 8-aligned token base
      const int iy = tb >> 3;
      const int h0 = (wy * kWS + iy + kShift) & (kH - 1);
#pragma unroll
      for (int q = 0; q < 2; ++q) {
        const int w0  = (wx * kWS + q * 4 + kShift) & (kW - 1);
        const int off = h0 * kW + w0;
        v4f xv = *(const v4f*)(xc + off);
        v4f res;
#pragma unroll
        for (int r = 0; r < 4; ++r) res[r] = acc[mt][q * 4 + r] + pb + xv[r];
        *(v4f*)(x1c + off) = res;
      }
    }
  }
}

// ---------------- kernel: fused LN2 + fc1 + GeLU + fc2 + residual -----------
// one workgroup (384 threads) per 64 consecutive pixels (never crosses a row)
__global__ __launch_bounds__(384)
void mlp_kernel(const float* __restrict__ x1,
                const float* __restrict__ n2w, const float* __restrict__ n2b,
                const __bf16* __restrict__ fc1w, const float* __restrict__ fc1_b,
                const __bf16* __restrict__ fc2w, const float* __restrict__ fc2_b,
                float* __restrict__ out) {
  extern __shared__ char smem[];
  __bf16* xn      = (__bf16*)(smem);           // 64x192 bf16  [0,24576)
  float*  x1stage = (float*)(smem + 24576);    // 192x64 f32 stage (in hh region)
  __bf16* hh      = (__bf16*)(smem + 24576);   // 64x768 bf16  [24576,122880)
  float*  red     = (float*)(smem + 122880);   // 2x64x6 f32

  const int tid = threadIdx.x;
  const int wv  = tid >> 5;
  const int t0  = tid / 6;
  const int g0  = tid % 6;
  const size_t pbase = (size_t)blockIdx.x * 64;
  const int bb   = (int)(pbase / kHW);
  const int rem0 = (int)(pbase % kHW);

  // ---- stage x1 tile (fully contiguous float4 loads) ----
#pragma unroll
  for (int it = 0; it < 8; ++it) {
    const int idx = it * 384 + tid;
    const int c   = idx >> 4;
    const int grp = idx & 15;
    v4f g = *(const v4f*)(x1 + ((size_t)bb * kC + c) * kHW + rem0 + grp * 4);
    *(v4f*)(x1stage + c * 64 + grp * 4) = g;
  }
  __syncthreads();

  // ---- LN2 from LDS stage ----
  {
    float vals[32];
    float s = 0.f, s2 = 0.f;
#pragma unroll
    for (int j = 0; j < 32; ++j) {
      float v = x1stage[(g0 * 32 + j) * 64 + t0];
      vals[j] = v; s += v; s2 += v * v;
    }
    red[t0 * 6 + g0]       = s;
    red[384 + t0 * 6 + g0] = s2;
    __syncthreads();
    float ts = 0.f, ts2 = 0.f;
#pragma unroll
    for (int g = 0; g < 6; ++g) { ts += red[t0 * 6 + g]; ts2 += red[384 + t0 * 6 + g]; }
    float mu   = ts / (float)kC;
    float var  = ts2 / (float)kC - mu * mu;
    float rstd = rsqrtf(var + 1e-5f);
#pragma unroll
    for (int j = 0; j < 32; j += 2) {
      int c = g0 * 32 + j;
      store_bf2(xn + t0 * kC + c,
                (vals[j]     - mu) * rstd * n2w[c]     + n2b[c],
                (vals[j + 1] - mu) * rstd * n2w[c + 1] + n2b[c + 1]);
    }
  }
  __syncthreads();

  // ---- fc1 (64x192x768) + GeLU -> hh bf16 ----
  {
    v8f acc[4][4];
#pragma unroll
    for (int mt = 0; mt < 4; ++mt)
#pragma unroll
      for (int nt = 0; nt < 4; ++nt) acc[mt][nt] = vzero8();
    for (int kk = 0; kk < 6; ++kk) {
      v16bf a[4];
#pragma unroll
      for (int mt = 0; mt < 4; ++mt) a[mt] = load_frag_a(xn + mt * 16 * kC + kk * 32, kC);
#pragma unroll
      for (int nt = 0; nt < 4; ++nt) {
        const int ncol = (wv * 4 + nt) * 16;
        v16bf bf = load_frag_b(fc1w + (size_t)ncol * kC + kk * 32, kC);
#pragma unroll
        for (int mt = 0; mt < 4; ++mt) acc[mt][nt] = wmma_bf16(a[mt], bf, acc[mt][nt]);
      }
    }
    const int lane = tid & 31;
    const int n    = lane & 15;
    const int mo   = (lane & 16) ? 8 : 0;
#pragma unroll
    for (int nt = 0; nt < 4; ++nt) {
      const int col = (wv * 4 + nt) * 16 + n;
      const float fb = fc1_b[col];
#pragma unroll
      for (int mt = 0; mt < 4; ++mt)
#pragma unroll
        for (int r = 0; r < 8; ++r) {
          float v = acc[mt][nt][r] + fb;
          v = 0.5f * v * (1.0f + erff(v * 0.70710678118f)); // exact GeLU
          hh[(mt * 16 + mo + r) * kHidden + col] = f2bf(v);
        }
    }
  }
  __syncthreads();

  // ---- fc2 (64x768x192) + residual -> out (float4 stores) ----
  {
    v8f acc[4];
#pragma unroll
    for (int mt = 0; mt < 4; ++mt) acc[mt] = vzero8();
    for (int kk = 0; kk < 24; ++kk) {
      v16bf bf = load_frag_b(fc2w + (size_t)(wv * 16) * kHidden + kk * 32, kHidden);
#pragma unroll
      for (int mt = 0; mt < 4; ++mt) {
        v16bf a = load_frag_a(hh + mt * 16 * kHidden + kk * 32, kHidden);
        acc[mt] = wmma_bf16(a, bf, acc[mt]);
      }
    }
    const int lane = tid & 31;
    const int n    = lane & 15;
    const int mo   = (lane & 16) ? 8 : 0;
    const int col  = wv * 16 + n;
    const float fb = fc2_b[col];
    const float* rc = x1  + ((size_t)bb * kC + col) * kHW + rem0;
    float*       oc = out + ((size_t)bb * kC + col) * kHW + rem0;
#pragma unroll
    for (int mt = 0; mt < 4; ++mt) {
      const int tb = mt * 16 + mo;
#pragma unroll
      for (int q = 0; q < 2; ++q) {
        v4f rv = *(const v4f*)(rc + tb + q * 4);
        v4f res;
#pragma unroll
        for (int r = 0; r < 4; ++r) res[r] = acc[mt][q * 4 + r] + fb + rv[r];
        *(v4f*)(oc + tb + q * 4) = res;
      }
    }
  }
}

// ---------------- launch ----------------
extern "C" void kernel_launch(void* const* d_in, const int* in_sizes, int n_in,
                              void* d_out, int out_size, void* d_ws, size_t ws_size,
                              hipStream_t stream) {
  (void)in_sizes; (void)n_in; (void)out_size; (void)ws_size;
  const float* x           = (const float*)d_in[0];
  const float* n1w         = (const float*)d_in[1];
  const float* n1b         = (const float*)d_in[2];
  const float* qkv_w       = (const float*)d_in[3];
  const float* q_bias      = (const float*)d_in[4];
  const float* v_bias      = (const float*)d_in[5];
  const float* logit_scale = (const float*)d_in[6];
  const float* cpb_w1      = (const float*)d_in[7];
  const float* cpb_b1      = (const float*)d_in[8];
  const float* cpb_w2      = (const float*)d_in[9];
  const float* proj_w      = (const float*)d_in[10];
  const float* proj_b      = (const float*)d_in[11];
  const float* n2w         = (const float*)d_in[12];
  const float* n2b         = (const float*)d_in[13];
  const float* fc1_w       = (const float*)d_in[14];
  const float* fc1_b       = (const float*)d_in[15];
  const float* fc2_w       = (const float*)d_in[16];
  const float* fc2_b       = (const float*)d_in[17];

  char* ws = (char*)d_ws;
  __bf16* qkvw   = (__bf16*)(ws + 0);        // 576*192*2  = 221184
  __bf16* projw  = (__bf16*)(ws + 221184);   // 192*192*2  =  73728
  __bf16* fc1w   = (__bf16*)(ws + 294912);   // 768*192*2  = 294912
  __bf16* fc2w   = (__bf16*)(ws + 589824);   // 192*768*2  = 294912
  float*  bias16 = (float*)(ws + 884736);    // 6*64*64*4  =  98304
  float*  x1     = (float*)(ws + 983040);    // 8*192*128*128*4 = 100663296

  prep_weights<<<256, 256, 0, stream>>>(qkv_w, proj_w, fc1_w, fc2_w,
                                        qkvw, projw, fc1w, fc2w);
  cpb_bias_kernel<<<1, 256, 0, stream>>>(cpb_w1, cpb_b1, cpb_w2, bias16);
  attn_kernel<<<kNwin, 384, 224256, stream>>>(x, n1w, n1b, qkvw, q_bias, v_bias,
                                              logit_scale, bias16, projw, proj_b, x1);
  mlp_kernel<<<(kB * kHW) / 64, 384, 125952, stream>>>(x1, n2w, n2b,
                                                       fc1w, fc1_b, fc2w, fc2_b,
                                                       (float*)d_out);
}
